// GroupedQueryAttention_65403761984312
// MI455X (gfx1250) — matrix operand order repaired and checked by reading, not compiled, not run on hardware
//
#include <hip/hip_runtime.h>
#include <hip/hip_bf16.h>
#include <stdint.h>

// Problem constants (from reference)
#define BB   2
#define TT   2048
#define DD   2048
#define HQn  32
#define HKVn 8
#define HDn  64
#define Gn   4
#define DKV  (HKVn * HDn)   // 512

typedef __bf16 bf16;
typedef bf16 v16bf __attribute__((ext_vector_type(16)));
typedef bf16 v8bf  __attribute__((ext_vector_type(8)));
typedef float v8f  __attribute__((ext_vector_type(8)));
typedef int   v4i  __attribute__((ext_vector_type(4)));

#define AS_GLOBAL __attribute__((address_space(1)))
#define AS_LOCAL  __attribute__((address_space(3)))

union FragAB {
    v16bf v;
    v8bf  h[2];
};

// ---- CDNA5 async memory->LDS path (guarded; falls back to load+ds_store)
#if defined(__has_builtin)
#if __has_builtin(__builtin_amdgcn_global_load_async_to_lds_b128)
#define HAVE_ASYNC_LDS 1
#endif
#endif
#ifndef HAVE_ASYNC_LDS
#define HAVE_ASYNC_LDS 0
#endif

__device__ inline void cp16_to_lds(const uint16_t* g, uint16_t* l) {
#if HAVE_ASYNC_LDS
    uint16_t* gnc = const_cast<uint16_t*>(g);
    __builtin_amdgcn_global_load_async_to_lds_b128(
        (AS_GLOBAL v4i*)gnc, (AS_LOCAL v4i*)l, 0, 0);
#else
    *(v8bf*)l = *(const v8bf*)g;
#endif
}

__device__ inline void wait_async_lds() {
#if HAVE_ASYNC_LDS
#if defined(__has_builtin) && __has_builtin(__builtin_amdgcn_s_wait_asynccnt)
    __builtin_amdgcn_s_wait_asynccnt(0);
#else
    asm volatile("s_wait_asynccnt 0" ::: "memory");
#endif
#endif
}

__device__ inline uint16_t f2bf(float f) {
    uint32_t u = __float_as_uint(f);
    uint32_t r = u + 0x7FFFu + ((u >> 16) & 1u);   // round-to-nearest-even
    return (uint16_t)(r >> 16);
}

__device__ inline v8f wmma_bf16(const FragAB& a, const FragAB& b, v8f c) {
    // D = A(16x32 bf16) x B(32x16 bf16) + C(16x16 f32)
    return __builtin_amdgcn_wmma_f32_16x16x32_bf16(
        false, a.v, false, b.v, (short)0, c, false, false);
}

// ---------------------------------------------------------------- converts
__global__ void k_f32_to_bf16(const float* __restrict__ src,
                              uint16_t* __restrict__ dst, int n) {
    int i = blockIdx.x * blockDim.x + threadIdx.x;
    if (i < n) dst[i] = f2bf(src[i]);
}

// ------------------------------------------------- block-tiled C = A*W^T + bias
// A: MxK bf16 row-major, W: NxK bf16 row-major, C: MxN fp32.
// Block (256 thr, 8 waves) computes 128x64; tiles staged in LDS, double buffered,
// filled with async global->LDS copies. Each wave computes a 32x32 sub-tile.
#define GBM 128
#define GBN 64
#define LSTRIDE 40        // LDS row stride in elements (80B, keeps 16B alignment)

__global__ __launch_bounds__(256) void k_gemm_bt_bf16(
    const uint16_t* __restrict__ A,
    const uint16_t* __restrict__ W,
    const float*    __restrict__ bias,
    float*          __restrict__ C,
    int M, int N, int K)
{
    __shared__ uint16_t sA[2][GBM * LSTRIDE];   // 2*128*40*2 = 20.0 KB
    __shared__ uint16_t sB[2][GBN * LSTRIDE];   // 2* 64*40*2 = 10.0 KB

    const int tid  = threadIdx.x;
    const int lane = tid & 31;
    const int wv   = tid >> 5;          // 0..7
    const int wm   = wv >> 1;           // 0..3 : m sub-tile
    const int wn   = wv & 1;            // 0..1 : n sub-tile
    const int half = lane >> 4, l15 = lane & 15;

    const int nblk = N / GBN;
    const int bm0  = (blockIdx.x / nblk) * GBM;
    const int bn0  = (blockIdx.x % nblk) * GBN;

    // stage one BK=32 slab of A (128x32) and W (64x32) into LDS buffer `buf`
    auto issueTile = [&](int buf, int k0) {
        const uint16_t* Ab = A + (size_t)bm0 * K + k0;
        const uint16_t* Wb = W + (size_t)bn0 * K + k0;
        // A: 512 16B-chunks (4 per row), threads take chunk tid and tid+256
        int c = tid;
        int row = c >> 2, col = (c & 3) * 8;
        cp16_to_lds(Ab + (size_t)row * K + col, &sA[buf][row * LSTRIDE + col]);
        c = tid + 256;
        row = c >> 2; col = (c & 3) * 8;
        cp16_to_lds(Ab + (size_t)row * K + col, &sA[buf][row * LSTRIDE + col]);
        // W: 256 16B-chunks (4 per row), one per thread
        row = tid >> 2; col = (tid & 3) * 8;
        cp16_to_lds(Wb + (size_t)row * K + col, &sB[buf][row * LSTRIDE + col]);
    };

    v8f acc00 = {}, acc01 = {}, acc10 = {}, acc11 = {};

    issueTile(0, 0);
    const int kSteps = K >> 5;
    for (int ks = 0; ks < kSteps; ++ks) {
        const int buf = ks & 1;
        wait_async_lds();
        __syncthreads();                 // all waves' tile data visible; prev readers done
        if (ks + 1 < kSteps) issueTile(buf ^ 1, (ks + 1) * 32);

        const uint16_t* aB  = &sA[buf][(wm * 32 + l15) * LSTRIDE + half * 16];
        const uint16_t* bBp = &sB[buf][(wn * 32 + l15) * LSTRIDE + half * 16];
        FragAB a0, a1, b0, b1;
        a0.h[0] = *(const v8bf*)(aB);
        a0.h[1] = *(const v8bf*)(aB + 8);
        a1.h[0] = *(const v8bf*)(aB + 16 * LSTRIDE);
        a1.h[1] = *(const v8bf*)(aB + 16 * LSTRIDE + 8);
        b0.h[0] = *(const v8bf*)(bBp);
        b0.h[1] = *(const v8bf*)(bBp + 8);
        b1.h[0] = *(const v8bf*)(bBp + 16 * LSTRIDE);
        b1.h[1] = *(const v8bf*)(bBp + 16 * LSTRIDE + 8);
        acc00 = wmma_bf16(a0, b0, acc00);
        acc01 = wmma_bf16(a0, b1, acc01);
        acc10 = wmma_bf16(a1, b0, acc10);
        acc11 = wmma_bf16(a1, b1, acc11);
    }

    const int m0 = bm0 + wm * 32;
    const int n0 = bn0 + wn * 32;
    const float bia0 = bias[n0 + l15];
    const float bia1 = bias[n0 + 16 + l15];
#pragma unroll
    for (int r = 0; r < 8; ++r) {
        const int row0 = m0 + r + half * 8;
        const int row1 = row0 + 16;
        C[(size_t)row0 * N + n0 + l15]      = acc00[r] + bia0;
        C[(size_t)row0 * N + n0 + 16 + l15] = acc01[r] + bia1;
        C[(size_t)row1 * N + n0 + l15]      = acc10[r] + bia0;
        C[(size_t)row1 * N + n0 + 16 + l15] = acc11[r] + bia1;
    }
}

// ------------------------------------------------- RoPE + layout repacks
// Q fp32 (B*T, D) -> Qb bf16 (B,HQ,T,HD), scaled by 1/sqrt(HD)
__global__ void k_rope_q(const float* __restrict__ Q, uint16_t* __restrict__ Qb) {
    int idx = blockIdx.x * blockDim.x + threadIdx.x;      // B*HQ*T*32 = 2^22
    if (idx >= BB * HQn * TT * (HDn / 2)) return;
    int i = idx & 31;
    int t = (idx >> 5) & (TT - 1);
    int h = (idx >> 16) & (HQn - 1);
    int b = idx >> 21;
    const float* src = Q + ((size_t)(b * TT + t)) * DD + h * HDn;
    float q1 = src[i], q2 = src[i + 32];
    float inv = __powf(10000.0f, -(float)(2 * i) / (float)HDn);
    float ang = (float)t * inv;
    float s, c;
    __sincosf(ang, &s, &c);
    uint16_t* dst = Qb + ((size_t)((b * HQn + h) * TT + t)) * HDn;
    dst[i]      = f2bf((q1 * c - q2 * s) * 0.125f);
    dst[i + 32] = f2bf((q2 * c + q1 * s) * 0.125f);
}

// K fp32 (B*T, DKV) -> Kb bf16 (B,HKV,T,HD)
__global__ void k_rope_k(const float* __restrict__ K, uint16_t* __restrict__ Kb) {
    int idx = blockIdx.x * blockDim.x + threadIdx.x;      // B*HKV*T*32 = 2^20
    if (idx >= BB * HKVn * TT * (HDn / 2)) return;
    int i = idx & 31;
    int t = (idx >> 5) & (TT - 1);
    int h = (idx >> 16) & (HKVn - 1);
    int b = idx >> 19;
    const float* src = K + ((size_t)(b * TT + t)) * DKV + h * HDn;
    float k1 = src[i], k2 = src[i + 32];
    float inv = __powf(10000.0f, -(float)(2 * i) / (float)HDn);
    float ang = (float)t * inv;
    float s, c;
    __sincosf(ang, &s, &c);
    uint16_t* dst = Kb + ((size_t)((b * HKVn + h) * TT + t)) * HDn;
    dst[i]      = f2bf(k1 * c - k2 * s);
    dst[i + 32] = f2bf(k2 * c + k1 * s);
}

// V fp32 (B*T, DKV) -> Vt bf16 (B,HKV,HD,T)  (transposed so PV B-frags are contiguous)
__global__ void k_pack_v(const float* __restrict__ V, uint16_t* __restrict__ Vt) {
    int idx = blockIdx.x * blockDim.x + threadIdx.x;      // B*HKV*T*HD = 2^21
    if (idx >= BB * HKVn * TT * HDn) return;
    int d = idx & 63;
    int t = (idx >> 6) & (TT - 1);
    int h = (idx >> 17) & (HKVn - 1);
    int b = idx >> 20;
    float v = V[((size_t)(b * TT + t)) * DKV + h * HDn + d];
    Vt[((size_t)((b * HKVn + h) * HDn + d)) * TT + t] = f2bf(v);
}

// ------------------------------------------------- flash attention, 1 wave / q-tile
__global__ __launch_bounds__(32) void k_attn(
    const uint16_t* __restrict__ Qb,   // (B,HQ,T,HD)
    const uint16_t* __restrict__ Kb,   // (B,HKV,T,HD)
    const uint16_t* __restrict__ Vt,   // (B,HKV,HD,T)
    uint16_t*       __restrict__ Ob)   // (B,T,D) bf16
{
    __shared__ float    sS[16][33];
    __shared__ uint16_t sP[16][32];
    __shared__ float    sAlpha[16];
    __shared__ float    sL[16];

    const int lane = threadIdx.x & 31;
    const int half = lane >> 4, l15 = lane & 15;
    const int blk = blockIdx.x;                // B*HQ*(T/16) = 8192
    const int qt = blk & (TT / 16 - 1);
    const int h  = (blk >> 7) & (HQn - 1);
    const int b  = blk >> 12;
    const int hk = h >> 2;                     // repeat_interleave: kv head = h / G
    const int t0 = qt * 16;

    // Q A-frags for hd 0..31 and 32..63
    const uint16_t* qrow =
        Qb + ((size_t)((b * HQn + h) * TT) + t0 + l15) * HDn + half * 16;
    FragAB qa0, qa1;
    qa0.h[0] = *(const v8bf*)(qrow);
    qa0.h[1] = *(const v8bf*)(qrow + 8);
    qa1.h[0] = *(const v8bf*)(qrow + 32);
    qa1.h[1] = *(const v8bf*)(qrow + 40);

    const uint16_t* kbase = Kb + ((size_t)(b * HKVn + hk) * TT) * HDn;
    const uint16_t* vbase = Vt + ((size_t)(b * HKVn + hk) * HDn) * TT;

    v8f acc0 = {}, acc1 = {}, acc2 = {}, acc3 = {};
    float m_i = -1e30f, l_i = 0.0f;

    for (int kb0 = 0; kb0 < TT; kb0 += 32) {
        // ---- S = Q * K^T  (16 x 32 scores)
        v8f s0 = {}, s1 = {};
        {
            const uint16_t* kr0 = kbase + (size_t)(kb0 + l15) * HDn + half * 16;
            const uint16_t* kr1 = kr0 + (size_t)16 * HDn;
            FragAB b00, b01, b10, b11;
            b00.h[0] = *(const v8bf*)(kr0);      b00.h[1] = *(const v8bf*)(kr0 + 8);
            b01.h[0] = *(const v8bf*)(kr0 + 32); b01.h[1] = *(const v8bf*)(kr0 + 40);
            b10.h[0] = *(const v8bf*)(kr1);      b10.h[1] = *(const v8bf*)(kr1 + 8);
            b11.h[0] = *(const v8bf*)(kr1 + 32); b11.h[1] = *(const v8bf*)(kr1 + 40);
            s0 = wmma_bf16(qa0, b00, s0);
            s0 = wmma_bf16(qa1, b01, s0);
            s1 = wmma_bf16(qa0, b10, s1);
            s1 = wmma_bf16(qa1, b11, s1);
        }
#pragma unroll
        for (int r = 0; r < 8; ++r) {
            sS[r + half * 8][l15]      = s0[r];
            sS[r + half * 8][l15 + 16] = s1[r];
        }
        __syncthreads();

        // ---- online softmax (lanes 0..15, one row each)
        if (lane < 16) {
            const int row = lane;
            float mx = m_i;
#pragma unroll
            for (int j = 0; j < 32; ++j) mx = fmaxf(mx, sS[row][j]);
            float alpha = __expf(m_i - mx);
            float sum = 0.0f;
#pragma unroll
            for (int j = 0; j < 32; ++j) {
                float p = __expf(sS[row][j] - mx);
                sP[row][j] = f2bf(p);
                sum += p;
            }
            l_i = l_i * alpha + sum;
            m_i = mx;
            sAlpha[row] = alpha;
        }
        __syncthreads();

        // ---- rescale accumulators
#pragma unroll
        for (int r = 0; r < 8; ++r) {
            float a = sAlpha[r + half * 8];
            acc0[r] *= a; acc1[r] *= a; acc2[r] *= a; acc3[r] *= a;
        }

        // ---- acc += P * V
        FragAB pa;
        pa.h[0] = *(const v8bf*)(&sP[l15][half * 16]);
        pa.h[1] = *(const v8bf*)(&sP[l15][half * 16 + 8]);
        const uint16_t* vr = vbase + (size_t)l15 * TT + kb0 + half * 16;
        FragAB v0, v1, v2, v3;
        v0.h[0] = *(const v8bf*)(vr);           v0.h[1] = *(const v8bf*)(vr + 8);
        v1.h[0] = *(const v8bf*)(vr + 16 * TT); v1.h[1] = *(const v8bf*)(vr + 16 * TT + 8);
        v2.h[0] = *(const v8bf*)(vr + 32 * TT); v2.h[1] = *(const v8bf*)(vr + 32 * TT + 8);
        v3.h[0] = *(const v8bf*)(vr + 48 * TT); v3.h[1] = *(const v8bf*)(vr + 48 * TT + 8);
        acc0 = wmma_bf16(pa, v0, acc0);
        acc1 = wmma_bf16(pa, v1, acc1);
        acc2 = wmma_bf16(pa, v2, acc2);
        acc3 = wmma_bf16(pa, v3, acc3);
        __syncthreads();
    }

    if (lane < 16) sL[lane] = l_i;
    __syncthreads();
#pragma unroll
    for (int r = 0; r < 8; ++r) {
        const int row = r + half * 8;
        const float inv = 1.0f / sL[row];
        uint16_t* dst =
            Ob + ((size_t)(b * TT + t0 + row)) * DD + h * HDn + l15;
        dst[0]  = f2bf(acc0[r] * inv);
        dst[16] = f2bf(acc1[r] * inv);
        dst[32] = f2bf(acc2[r] * inv);
        dst[48] = f2bf(acc3[r] * inv);
    }
}

// ----------------------------------------------------------------- launcher
extern "C" void kernel_launch(void* const* d_in, const int* in_sizes, int n_in,
                              void* d_out, int out_size, void* d_ws, size_t ws_size,
                              hipStream_t stream) {
    (void)in_sizes; (void)n_in; (void)out_size; (void)ws_size;
    const float* x  = (const float*)d_in[0];
    const float* Wq = (const float*)d_in[1];
    const float* bq = (const float*)d_in[2];
    const float* Wk = (const float*)d_in[3];
    const float* bk = (const float*)d_in[4];
    const float* Wv = (const float*)d_in[5];
    const float* bv = (const float*)d_in[6];
    const float* Wo = (const float*)d_in[7];
    const float* bo = (const float*)d_in[8];
    float* out = (float*)d_out;

    char* ws = (char*)d_ws;
    size_t off = 0;
    auto salloc = [&](size_t bytes) -> void* {
        void* p = ws + off;
        off += (bytes + 255) & ~(size_t)255;
        return p;
    };

    const int nX  = BB * TT * DD;        // 8388608
    const int nWq = DD * DD;             // 4194304
    const int nWk = DKV * DD;            // 1048576
    const int nKV = BB * TT * DKV;       // 2097152

    uint16_t* xb  = (uint16_t*)salloc((size_t)nX  * 2);
    uint16_t* wqb = (uint16_t*)salloc((size_t)nWq * 2);
    uint16_t* wkb = (uint16_t*)salloc((size_t)nWk * 2);
    uint16_t* wvb = (uint16_t*)salloc((size_t)nWk * 2);
    uint16_t* wob = (uint16_t*)salloc((size_t)nWq * 2);
    float*    qf  = (float*)   salloc((size_t)nX  * 4);
    float*    kf  = (float*)   salloc((size_t)nKV * 4);
    float*    vf  = (float*)   salloc((size_t)nKV * 4);
    uint16_t* qb  = (uint16_t*)salloc((size_t)nX  * 2);
    uint16_t* kb  = (uint16_t*)salloc((size_t)nKV * 2);
    uint16_t* vt  = (uint16_t*)salloc((size_t)nKV * 2);
    uint16_t* ab  = (uint16_t*)salloc((size_t)nX  * 2);

    // 1) fp32 -> bf16
    k_f32_to_bf16<<<(nX  + 255) / 256, 256, 0, stream>>>(x,  xb,  nX);
    k_f32_to_bf16<<<(nWq + 255) / 256, 256, 0, stream>>>(Wq, wqb, nWq);
    k_f32_to_bf16<<<(nWk + 255) / 256, 256, 0, stream>>>(Wk, wkb, nWk);
    k_f32_to_bf16<<<(nWk + 255) / 256, 256, 0, stream>>>(Wv, wvb, nWk);
    k_f32_to_bf16<<<(nWq + 255) / 256, 256, 0, stream>>>(Wo, wob, nWq);

    const int M = BB * TT;               // 4096
    // 2) projections (block-tiled LDS WMMA GEMMs): grid = (M/128)*(N/64)
    k_gemm_bt_bf16<<<(M / GBM) * (DD / GBN), 256, 0, stream>>>(
        xb, wqb, bq, qf, M, DD, DD);
    k_gemm_bt_bf16<<<(M / GBM) * (DKV / GBN), 256, 0, stream>>>(
        xb, wkb, bk, kf, M, DKV, DD);
    k_gemm_bt_bf16<<<(M / GBM) * (DKV / GBN), 256, 0, stream>>>(
        xb, wvb, bv, vf, M, DKV, DD);

    // 3) RoPE + repack to head-major bf16 (V transposed)
    k_rope_q<<<(BB * HQn * TT * 32 + 255) / 256, 256, 0, stream>>>(qf, qb);
    k_rope_k<<<(BB * HKVn * TT * 32 + 255) / 256, 256, 0, stream>>>(kf, kb);
    k_pack_v<<<(BB * HKVn * TT * HDn + 255) / 256, 256, 0, stream>>>(vf, vt);

    // 4) flash attention (WMMA), one wave per (b, h, 16-query tile)
    k_attn<<<BB * HQn * (TT / 16), 32, 0, stream>>>(qb, kb, vt, ab);

    // 5) output projection -> fp32 result
    k_gemm_bt_bf16<<<(M / GBM) * (DD / GBN), 256, 0, stream>>>(
        ab, wob, bo, out, M, DD, DD);
}
